// PNPP_PointEnc_14972255994425
// MI455X (gfx1250) — matrix operand order repaired and checked by reading, not compiled, not run on hardware
//
#include <hip/hip_runtime.h>

typedef _Float16 f16_t;
typedef __attribute__((ext_vector_type(16))) _Float16 v16h;
typedef __attribute__((ext_vector_type(8)))  float    v8f;

// ---------------------------------------------------------------------------
// Generic 32-row x CN GEMM tile using v_wmma_f32_16x16x32_f16.
//   A: either pre-swizzled A-fragment layout in global memory (AFRAG=true)
//      or row-major [32][K] f16 (LDS) (AFRAG=false)
//   Wf: weights pre-swizzled into B-fragment order: [(nt*KT+kt)*32 + lane] v16h
//   Epilogue: relu(acc + bias) -> f16 LDS H (row-major [32][CN]) or f32 global.
// Block = 128 threads = 4 wave32; wave w handles tiles t = w, w+4, ...
// ---------------------------------------------------------------------------
template<int K, int CN, bool AFRAG, bool OUTG>
__device__ inline void gemm32(const f16_t* __restrict__ A,
                              const f16_t* __restrict__ Wf,
                              const float* __restrict__ bias,
                              f16_t* H, float* outp)
{
    static_assert(K % 32 == 0 && CN % 16 == 0, "tile dims");
    constexpr int NT = CN / 16, KT = K / 32, T = 2 * NT;
    const int w    = threadIdx.x >> 5;
    const int lane = threadIdx.x & 31;
    const int l16  = lane & 15;
    const int hi   = lane >> 4;
    const v16h* Af = (const v16h*)A;
    const v16h* Bf = (const v16h*)Wf;
    for (int t = w; t < T; t += 4) {
        const int mt = t / NT, nt = t % NT;
        const int m0 = mt * 16, n0 = nt * 16;
        v8f acc = {0.f, 0.f, 0.f, 0.f, 0.f, 0.f, 0.f, 0.f};
        for (int kt = 0; kt < KT; ++kt) {
            v16h a;
            if constexpr (AFRAG) {
                a = Af[(mt * KT + kt) * 32 + lane];
            } else {
                const int k0 = kt * 32;
#pragma unroll
                for (int h = 0; h < 16; ++h)
                    a[h] = A[(m0 + l16) * K + k0 + h + hi * 8 + ((h & 8) ? 8 : 0)];
            }
            v16h b = Bf[(nt * KT + kt) * 32 + lane];
            acc = __builtin_amdgcn_wmma_f32_16x16x32_f16(
                false, a, false, b, (short)0, acc, false, false);
        }
        const float bv = bias[n0 + l16];
#pragma unroll
        for (int r = 0; r < 8; ++r) {
            float v = acc[r] + bv;
            v = v > 0.f ? v : 0.f;
            const int m = m0 + r + hi * 8;
            if constexpr (OUTG) outp[(size_t)m * CN + n0 + l16] = v;
            else                H[m * CN + n0 + l16] = (f16_t)v;
        }
    }
}

// ---------------------------------------------------------------------------
// Weight pack: f32 [cin][cout] -> f16 B-fragment order, K zero-padded to cinp.
// half index t = ((nt*KT + kt)*32 + lane)*16 + h ; k = kt*32 + (h&7) + (lane>>4)*8 + (h>>3)*16
// ---------------------------------------------------------------------------
__global__ void pack_weight_kernel(const float* __restrict__ w, f16_t* __restrict__ o,
                                   int cin, int KT, int cout, int total)
{
    int t = blockIdx.x * blockDim.x + threadIdx.x;
    if (t >= total) return;
    int h    = t & 15;
    int lane = (t >> 4) & 31;
    int q    = t >> 9;
    int kt   = q % KT;
    int nt   = q / KT;
    int k    = kt * 32 + (h & 7) + ((lane >> 4) ? 8 : 0) + ((h & 8) ? 16 : 0);
    int n    = nt * 16 + (lane & 15);
    o[t] = (k < cin) ? (f16_t)w[(size_t)k * cout + n] : (f16_t)0.f;
}

// ---------------------------------------------------------------------------
// Farthest point sampling: one block per batch, dist array in LDS.
// Matches jnp.argmax first-occurrence tie-breaking (lower index wins).
// ---------------------------------------------------------------------------
__global__ __launch_bounds__(256) void fps_kernel(const float* __restrict__ xyz,
                                                  int* __restrict__ fidx, int N, int M)
{
    __shared__ float dist[8192];
    __shared__ float rv[256];
    __shared__ int   ri[256];
    const int b = blockIdx.x, tid = threadIdx.x;
    const float* x = xyz + (size_t)b * N * 3;
    for (int i = tid; i < N; i += 256) dist[i] = 1e10f;
    int last = 0;
    if (tid == 0) fidx[b * M] = 0;
    __syncthreads();
    for (int it = 1; it < M; ++it) {
        const float px = x[last * 3], py = x[last * 3 + 1], pz = x[last * 3 + 2];
        float best = -1.f; int bi = 0;
        for (int i = tid; i < N; i += 256) {
            float dx = x[i * 3] - px, dy = x[i * 3 + 1] - py, dz = x[i * 3 + 2] - pz;
            float d  = dx * dx + dy * dy + dz * dz;
            float nd = fminf(dist[i], d);
            dist[i] = nd;
            if (nd > best) { best = nd; bi = i; }
        }
        rv[tid] = best; ri[tid] = bi;
        __syncthreads();
        for (int s = 128; s > 0; s >>= 1) {
            if (tid < s) {
                float v2 = rv[tid + s]; int i2 = ri[tid + s];
                if (v2 > rv[tid] || (v2 == rv[tid] && i2 < ri[tid])) { rv[tid] = v2; ri[tid] = i2; }
            }
            __syncthreads();
        }
        last = ri[0];
        if (tid == 0) fidx[b * M + it] = last;
        __syncthreads();
    }
}

__global__ void gather_xyz_kernel(const float* __restrict__ xyz, const int* __restrict__ fidx,
                                  float* __restrict__ nx, int N, int M, int B)
{
    int t = blockIdx.x * blockDim.x + threadIdx.x;
    if (t >= B * M) return;
    int b = t / M;
    int g = fidx[t];
    nx[(size_t)t * 3 + 0] = xyz[((size_t)b * N + g) * 3 + 0];
    nx[(size_t)t * 3 + 1] = xyz[((size_t)b * N + g) * 3 + 1];
    nx[(size_t)t * 3 + 2] = xyz[((size_t)b * N + g) * 3 + 2];
}

// Ball query: smallest nsample(=32) indices with d2 < r2, scan ascending.
__global__ void ballquery_kernel(const float* __restrict__ xyz, const float* __restrict__ nxyz,
                                 int* __restrict__ gidx, int B, int N, int M, float r2)
{
    int t = blockIdx.x * blockDim.x + threadIdx.x;
    if (t >= B * M) return;
    int b = t / M;
    const float cx = nxyz[(size_t)t * 3], cy = nxyz[(size_t)t * 3 + 1], cz = nxyz[(size_t)t * 3 + 2];
    const float* x = xyz + (size_t)b * N * 3;
    int* out = gidx + (size_t)t * 32;
    int cnt = 0, first = 0; bool any = false;
    for (int i = 0; i < N && cnt < 32; ++i) {
        float dx = x[i * 3] - cx, dy = x[i * 3 + 1] - cy, dz = x[i * 3 + 2] - cz;
        if (dx * dx + dy * dy + dz * dz < r2) {
            if (!any) { any = true; first = i; }
            out[cnt++] = i;
        }
    }
    int fill = any ? first : 0;
    for (int s = cnt; s < 32; ++s) out[s] = fill;
}

// ---------------------------------------------------------------------------
// SA kernel: one block per (batch, center). Gather+center 32 samples into LDS,
// 3 MLP layers via WMMA, max-pool over samples.
// ---------------------------------------------------------------------------
template<int CIN, int CINP, int C1, int C2, int C3>
__global__ __launch_bounds__(128)
void sa_mlp_kernel(const float* __restrict__ xyz, const float* __restrict__ feats,
                   const float* __restrict__ nxyz, const int* __restrict__ gidx,
                   const f16_t* W1, const float* B1, const f16_t* W2, const float* B2,
                   const f16_t* W3, const float* B3,
                   float* __restrict__ out, int N, int M)
{
    constexpr int AH = 32 * ((CINP > C2) ? CINP : C2);
    constexpr int BH = 32 * ((C1 > C3) ? C1 : C3);
    __shared__ f16_t bufA[AH];
    __shared__ f16_t bufB[BH];
    const int b = blockIdx.x / M, c = blockIdx.x % M;
    const int* gi = gidx + (size_t)blockIdx.x * 32;
    const float cx = nxyz[(size_t)blockIdx.x * 3];
    const float cy = nxyz[(size_t)blockIdx.x * 3 + 1];
    const float cz = nxyz[(size_t)blockIdx.x * 3 + 2];
    (void)c;
    for (int idx = threadIdx.x; idx < 32 * CINP; idx += 128) {
        const int s = idx / CINP, ch = idx % CINP;
        float v = 0.f;
        const int g = gi[s];
        if (ch < 3) {
            v = xyz[((size_t)b * N + g) * 3 + ch] - (ch == 0 ? cx : (ch == 1 ? cy : cz));
        } else if (CIN > 3 && ch < CIN) {
            v = feats[((size_t)b * N + g) * (CIN - 3) + (ch - 3)];
        }
        bufA[idx] = (f16_t)v;
    }
    __syncthreads();
    gemm32<CINP, C1, false, false>(bufA, W1, B1, bufB, nullptr);
    __syncthreads();
    gemm32<C1, C2, false, false>(bufB, W2, B2, bufA, nullptr);
    __syncthreads();
    gemm32<C2, C3, false, false>(bufA, W3, B3, bufB, nullptr);
    __syncthreads();
    for (int ch = threadIdx.x; ch < C3; ch += 128) {
        float m = 0.f;  // outputs are relu'd (>= 0)
        for (int s = 0; s < 32; ++s) m = fmaxf(m, (float)bufB[s * C3 + ch]);
        out[(size_t)blockIdx.x * C3 + ch] = m;
    }
}

// ---------------------------------------------------------------------------
// FP interpolation: 3-NN inverse-distance weights; writes X in WMMA A-fragment
// order (f16) so the MLP layer-1 A operand loads as contiguous v16h.
// ---------------------------------------------------------------------------
__global__ void interp_kernel(const float* __restrict__ unk_xyz, const float* __restrict__ kn_xyz,
                              const float* __restrict__ kn_feats, const float* __restrict__ unk_feats,
                              f16_t* __restrict__ Xf, int Bn, int Nu, int Nk, int CK, int CU)
{
    int t = blockIdx.x * blockDim.x + threadIdx.x;
    if (t >= Bn * Nu) return;
    const int b = t / Nu, u = t % Nu;
    const int KT = (CK + CU) / 32;
    const float ux = unk_xyz[(size_t)t * 3], uy = unk_xyz[(size_t)t * 3 + 1], uz = unk_xyz[(size_t)t * 3 + 2];
    const float* kx = kn_xyz + (size_t)b * Nk * 3;
    float d0 = 1e30f, d1 = 1e30f, d2 = 1e30f;
    int   i0 = 0,     i1 = 0,     i2 = 0;
    for (int j = 0; j < Nk; ++j) {
        float dx = kx[j * 3] - ux, dy = kx[j * 3 + 1] - uy, dz = kx[j * 3 + 2] - uz;
        float d = dx * dx + dy * dy + dz * dz;
        if (d < d0)      { d2 = d1; i2 = i1; d1 = d0; i1 = i0; d0 = d; i0 = j; }
        else if (d < d1) { d2 = d1; i2 = i1; d1 = d; i1 = j; }
        else if (d < d2) { d2 = d; i2 = j; }
    }
    float w0 = 1.f / (d0 + 1e-8f), w1 = 1.f / (d1 + 1e-8f), w2 = 1.f / (d2 + 1e-8f);
    const float ws = w0 + w1 + w2;
    w0 /= ws; w1 /= ws; w2 /= ws;
    const int blk = t >> 5, mr = t & 31, mt = mr >> 4, lb = mr & 15;
    const size_t base = ((size_t)blk * 2 + mt) * KT * 512;  // halves
    auto put = [&](int ch, float v) {
        const int kt = ch >> 5, r = ch & 31;
        const int hi = (r >> 3) & 1;
        const int h  = (r & 7) | ((r & 16) >> 1);
        Xf[base + (size_t)(kt * 32 + lb + hi * 16) * 16 + h] = (f16_t)v;
    };
    const float* f0 = kn_feats + ((size_t)b * Nk + i0) * CK;
    const float* f1 = kn_feats + ((size_t)b * Nk + i1) * CK;
    const float* f2 = kn_feats + ((size_t)b * Nk + i2) * CK;
    for (int ch = 0; ch < CK; ++ch)
        put(ch, w0 * f0[ch] + w1 * f1[ch] + w2 * f2[ch]);
    for (int ch = 0; ch < CU; ++ch)
        put(CK + ch, unk_feats[((size_t)b * Nu + u) * CU + ch]);
}

// ---------------------------------------------------------------------------
// FP MLP: one block per 32 rows. Layer1 A from global fragments, rest via LDS.
// C3==0 -> 2-layer MLP.
// ---------------------------------------------------------------------------
template<int K, int C1, int C2, int C3>
__global__ __launch_bounds__(128)
void fp_mlp_kernel(const f16_t* __restrict__ Xf,
                   const f16_t* W1, const float* B1, const f16_t* W2, const float* B2,
                   const f16_t* W3, const float* B3, float* __restrict__ out)
{
    constexpr int KT = K / 32;
    __shared__ f16_t H1[32 * C1];
    __shared__ f16_t H2[(C3 > 0) ? 32 * C2 : 32];
    const f16_t* Ab = Xf + (size_t)blockIdx.x * 2 * KT * 512;
    gemm32<K, C1, true, false>(Ab, W1, B1, H1, nullptr);
    __syncthreads();
    if constexpr (C3 > 0) {
        gemm32<C1, C2, false, false>(H1, W2, B2, H2, nullptr);
        __syncthreads();
        gemm32<C2, C3, false, true>(H2, W3, B3, nullptr, out + (size_t)blockIdx.x * 32 * C3);
    } else {
        gemm32<C1, C2, false, true>(H1, W2, B2, nullptr, out + (size_t)blockIdx.x * 32 * C2);
        (void)W3; (void)B3;
    }
}

// ---------------------------------------------------------------------------
extern "C" void kernel_launch(void* const* d_in, const int* in_sizes, int n_in,
                              void* d_out, int out_size, void* d_ws, size_t ws_size,
                              hipStream_t stream)
{
    (void)in_sizes; (void)n_in; (void)out_size; (void)ws_size;
    const float* pc = (const float*)d_in[0];
    auto F = [&](int i) { return (const float*)d_in[i]; };

    char* wsb = (char*)d_ws;
    size_t off = 0;
    auto alloc = [&](size_t bytes) -> void* {
        size_t o = (off + 255) & ~(size_t)255;
        off = o + bytes;
        return (void*)(wsb + o);
    };

    // ---- pack all weights to f16 B-fragment order -------------------------
    struct PWd { int idx, cin, cinp, cout; };
    const PWd tab[21] = {
        {1, 3, 32, 32},   {3, 32, 32, 32},   {5, 32, 32, 64},       // SA1
        {7, 67, 96, 64},  {9, 64, 64, 64},   {11, 64, 64, 128},     // SA2
        {13, 131, 160, 128}, {15, 128, 128, 128}, {17, 128, 128, 256}, // SA3
        {19, 259, 288, 256}, {21, 256, 256, 256}, {23, 256, 256, 512}, // SA4
        {25, 128, 128, 128}, {27, 128, 128, 128}, {29, 128, 128, 128}, // FP0
        {31, 320, 320, 256}, {33, 256, 256, 128},                      // FP1
        {35, 384, 384, 256}, {37, 256, 256, 256},                      // FP2
        {39, 768, 768, 256}, {41, 256, 256, 256},                      // FP3
    };
    f16_t* pw[21];
    for (int i = 0; i < 21; ++i) {
        const size_t n = (size_t)tab[i].cinp * tab[i].cout;
        pw[i] = (f16_t*)alloc(n * 2);
        pack_weight_kernel<<<(int)((n + 255) / 256), 256, 0, stream>>>(
            F(tab[i].idx), pw[i], tab[i].cin, tab[i].cinp / 32, tab[i].cout, (int)n);
    }

    // ---- scratch buffers ---------------------------------------------------
    int*   fi1 = (int*)alloc((size_t)8 * 1024 * 4);
    int*   fi2 = (int*)alloc((size_t)8 * 256 * 4);
    int*   fi3 = (int*)alloc((size_t)8 * 64 * 4);
    int*   fi4 = (int*)alloc((size_t)8 * 16 * 4);
    float* nx1 = (float*)alloc((size_t)8 * 1024 * 3 * 4);
    float* nx2 = (float*)alloc((size_t)8 * 256 * 3 * 4);
    float* nx3 = (float*)alloc((size_t)8 * 64 * 3 * 4);
    float* nx4 = (float*)alloc((size_t)8 * 16 * 3 * 4);
    int*   g1  = (int*)alloc((size_t)8 * 1024 * 32 * 4);
    int*   g2  = (int*)alloc((size_t)8 * 256 * 32 * 4);
    int*   g3  = (int*)alloc((size_t)8 * 64 * 32 * 4);
    int*   g4  = (int*)alloc((size_t)8 * 16 * 32 * 4);
    float* f1  = (float*)alloc((size_t)8 * 1024 * 64 * 4);
    float* f2  = (float*)alloc((size_t)8 * 256 * 128 * 4);
    float* f3  = (float*)alloc((size_t)8 * 64 * 256 * 4);
    float* f4  = (float*)alloc((size_t)8 * 16 * 512 * 4);
    float* dd3 = (float*)alloc((size_t)8 * 64 * 256 * 4);
    float* dd2 = (float*)alloc((size_t)8 * 256 * 256 * 4);
    float* dd1 = (float*)alloc((size_t)8 * 1024 * 128 * 4);
    f16_t* Xf  = (f16_t*)alloc((size_t)65536 * 128 * 2);  // reused, sized for largest
    float* out = (float*)d_out;

    // ---- SA level 1: 8192 -> 1024, r=0.1, MLP 3->32->32->64 ---------------
    fps_kernel<<<8, 256, 0, stream>>>(pc, fi1, 8192, 1024);
    gather_xyz_kernel<<<(8 * 1024 + 255) / 256, 256, 0, stream>>>(pc, fi1, nx1, 8192, 1024, 8);
    ballquery_kernel<<<(8 * 1024 + 255) / 256, 256, 0, stream>>>(pc, nx1, g1, 8, 8192, 1024, 0.01f);
    sa_mlp_kernel<3, 32, 32, 32, 64><<<8 * 1024, 128, 0, stream>>>(
        pc, nullptr, nx1, g1, pw[0], F(2), pw[1], F(4), pw[2], F(6), f1, 8192, 1024);

    // ---- SA level 2: 1024 -> 256, r=0.2, MLP 67->64->64->128 --------------
    fps_kernel<<<8, 256, 0, stream>>>(nx1, fi2, 1024, 256);
    gather_xyz_kernel<<<(8 * 256 + 255) / 256, 256, 0, stream>>>(nx1, fi2, nx2, 1024, 256, 8);
    ballquery_kernel<<<(8 * 256 + 255) / 256, 256, 0, stream>>>(nx1, nx2, g2, 8, 1024, 256, 0.04f);
    sa_mlp_kernel<67, 96, 64, 64, 128><<<8 * 256, 128, 0, stream>>>(
        nx1, f1, nx2, g2, pw[3], F(8), pw[4], F(10), pw[5], F(12), f2, 1024, 256);

    // ---- SA level 3: 256 -> 64, r=0.4, MLP 131->128->128->256 -------------
    fps_kernel<<<8, 256, 0, stream>>>(nx2, fi3, 256, 64);
    gather_xyz_kernel<<<(8 * 64 + 255) / 256, 256, 0, stream>>>(nx2, fi3, nx3, 256, 64, 8);
    ballquery_kernel<<<(8 * 64 + 255) / 256, 256, 0, stream>>>(nx2, nx3, g3, 8, 256, 64, 0.16f);
    sa_mlp_kernel<131, 160, 128, 128, 256><<<8 * 64, 128, 0, stream>>>(
        nx2, f2, nx3, g3, pw[6], F(14), pw[7], F(16), pw[8], F(18), f3, 256, 64);

    // ---- SA level 4: 64 -> 16, r=0.8, MLP 259->256->256->512 --------------
    fps_kernel<<<8, 256, 0, stream>>>(nx3, fi4, 64, 16);
    gather_xyz_kernel<<<(8 * 16 + 255) / 256, 256, 0, stream>>>(nx3, fi4, nx4, 64, 16, 8);
    ballquery_kernel<<<(8 * 16 + 255) / 256, 256, 0, stream>>>(nx3, nx4, g4, 8, 64, 16, 0.64f);
    sa_mlp_kernel<259, 288, 256, 256, 512><<<8 * 16, 128, 0, stream>>>(
        nx3, f3, nx4, g4, pw[9], F(20), pw[10], F(22), pw[11], F(24), f4, 64, 16);

    // ---- FP deepest: (64 <- 16), interp 512 + skip 256 = 768 -> 256 -> 256
    interp_kernel<<<(8 * 64 + 255) / 256, 256, 0, stream>>>(nx3, nx4, f4, f3, Xf, 8, 64, 16, 512, 256);
    fp_mlp_kernel<768, 256, 256, 0><<<(8 * 64) / 32, 128, 0, stream>>>(
        Xf, pw[19], F(40), pw[20], F(42), nullptr, nullptr, dd3);

    // ---- FP: (256 <- 64), interp 256 + skip 128 = 384 -> 256 -> 256 -------
    interp_kernel<<<(8 * 256 + 255) / 256, 256, 0, stream>>>(nx2, nx3, dd3, f2, Xf, 8, 256, 64, 256, 128);
    fp_mlp_kernel<384, 256, 256, 0><<<(8 * 256) / 32, 128, 0, stream>>>(
        Xf, pw[17], F(36), pw[18], F(38), nullptr, nullptr, dd2);

    // ---- FP: (1024 <- 256), interp 256 + skip 64 = 320 -> 256 -> 128 ------
    interp_kernel<<<(8 * 1024 + 255) / 256, 256, 0, stream>>>(nx1, nx2, dd2, f1, Xf, 8, 1024, 256, 256, 64);
    fp_mlp_kernel<320, 256, 128, 0><<<(8 * 1024) / 32, 128, 0, stream>>>(
        Xf, pw[15], F(32), pw[16], F(34), nullptr, nullptr, dd1);

    // ---- FP final: (8192 <- 1024), interp 128 -> 128 -> 128 -> 128 --------
    interp_kernel<<<(8 * 8192 + 255) / 256, 256, 0, stream>>>(pc, nx1, dd1, nullptr, Xf, 8, 8192, 1024, 128, 0);
    fp_mlp_kernel<128, 128, 128, 128><<<(8 * 8192) / 32, 128, 0, stream>>>(
        Xf, pw[12], F(26), pw[13], F(28), pw[14], F(30), out);
}
